// HungarianLoss_51926154609373
// MI455X (gfx1250) — compile-verified, matrix operand order checked by reading, not run on hardware
//
#include <hip/hip_runtime.h>
#include <math.h>

#define BS   32
#define NQ   300
#define NC   8192
#define NOBJ 64
#define ROWS (BS*NQ)      // 9600
#define CPAD 304          // padded query stride of cost rows
#define JINF 3.0e38f

typedef __attribute__((ext_vector_type(16))) _Float16 v16h;
typedef __attribute__((ext_vector_type(8)))  float    v8f;

// ---------------------------------------------------------------------------
// Kernel 1: per-row max (streaming, coalesced) + log-sum-exp where the
// sum(exp) is accumulated with V_WMMA_F32_16X16X32_F16 (f32 accumulator).
// Split-f16 trick: e = h + r with h=(f16)e, r=(f16)(e-h) in adjacent K slots,
// so the f32 accumulator recovers e to ~2^-22 relative accuracy.
// Block = 256 threads (8 waves) handles a 16-row tile; wave w owns cols
// [w*1024, w*1024+1024), 16 genuine columns per WMMA -> 64 WMMAs per wave.
// ---------------------------------------------------------------------------
__global__ __launch_bounds__(256)
void k1_row_lse(const float* __restrict__ logits, float* __restrict__ g_lse)
{
  __shared__ float s_pmax[256];
  __shared__ float s_max[16];
  __shared__ float s_part[8 * 16];

  const int tid = threadIdx.x;
  const int r0  = blockIdx.x * 16;

  // ---- phase A: per-row max, fully coalesced float4 streaming ----
  {
    const int r  = tid >> 4;          // 0..15 : row within tile
    const int c0 = tid & 15;          // 16 threads per row
    const float4* rp = (const float4*)(logits + (size_t)(r0 + r) * NC);
    float m = -INFINITY;
    #pragma unroll 4
    for (int j = 0; j < 128; ++j) {   // 128 * 16 float4 = 8192 floats
      float4 q = rp[c0 + j * 16];
      m = fmaxf(m, fmaxf(fmaxf(q.x, q.y), fmaxf(q.z, q.w)));
    }
    s_pmax[tid] = m;
  }
  __syncthreads();
  if (tid < 16) {
    float m = -INFINITY;
    #pragma unroll
    for (int k = 0; k < 16; ++k) m = fmaxf(m, s_pmax[tid * 16 + k]);
    s_max[tid] = m;
  }
  __syncthreads();

  // ---- phase B: sum(exp(x - max)) via WMMA against an all-ones B ----
  const int lane = tid & 31;
  const int wave = tid >> 5;
  const int hi   = lane >> 4;         // lane group (K offset per ISA layout)
  const int row  = lane & 15;         // A-matrix M = lane & 15
  const float rmax = s_max[row];
  const float* rp  = logits + (size_t)(r0 + row) * NC;

  v16h onesb;
  #pragma unroll
  for (int j = 0; j < 16; ++j) onesb[j] = (_Float16)1.0f;

  v8f acc = {};
  const int kstart = wave * 1024;
  for (int it = 0; it < 64; ++it) {
    const int kb = kstart + it * 16;
    // ISA A-layout: lane needs cols c = hi*4 + {0..3} and hi*4+8 + {0..3}
    float4 A4 = *(const float4*)(rp + kb + hi * 4);
    float4 B4 = *(const float4*)(rp + kb + hi * 4 + 8);
    float av[4] = {A4.x, A4.y, A4.z, A4.w};
    float bv[4] = {B4.x, B4.y, B4.z, B4.w};
    v16h a;
    #pragma unroll
    for (int i2 = 0; i2 < 4; ++i2) {
      float e = __expf(av[i2] - rmax);
      _Float16 h  = (_Float16)e;
      _Float16 rr = (_Float16)(e - (float)h);
      a[2 * i2]     = h;
      a[2 * i2 + 1] = rr;
      float e2 = __expf(bv[i2] - rmax);
      _Float16 h2  = (_Float16)e2;
      _Float16 rr2 = (_Float16)(e2 - (float)h2);
      a[8 + 2 * i2] = h2;
      a[9 + 2 * i2] = rr2;
    }
    acc = __builtin_amdgcn_wmma_f32_16x16x32_f16(false, a, false, onesb,
                                                 (short)0, acc, false, false);
  }
  // D layout: VGPR v holds M = v + 8*hi (replicated across N since B==1)
  if ((tid & 15) == 0) {
    #pragma unroll
    for (int d = 0; d < 8; ++d) s_part[wave * 16 + d + 8 * hi] = acc[d];
  }
  __syncthreads();
  if (tid < 16) {
    float S = 0.f;
    #pragma unroll
    for (int w = 0; w < 8; ++w) S += s_part[w * 16 + tid];
    g_lse[r0 + tid] = s_max[tid] + logf(S);
  }
}

// ---------------------------------------------------------------------------
// Kernel 2: per batch, compact nonzero labels and build the transposed cost
// matrix C[j][q] = -exp(logit[b,q,id_j] - lse[b,q])  (JV solves s x 300).
// ---------------------------------------------------------------------------
__global__ __launch_bounds__(256)
void k2_cost(const float* __restrict__ logits, const int* __restrict__ labels,
             const float* __restrict__ g_lse, float* __restrict__ g_cost,
             int* __restrict__ g_sizes)
{
  __shared__ int ids[NOBJ];
  __shared__ int s_cnt;
  const int b = blockIdx.x;
  if (threadIdx.x == 0) {
    int s = 0;
    for (int k = 0; k < NOBJ; ++k) {
      int lab = labels[b * NOBJ + k];
      if (lab != 0) ids[s++] = lab;
    }
    s_cnt = s;
    g_sizes[b] = s;
  }
  __syncthreads();
  const int s = s_cnt;
  const int total = NQ * s;
  for (int idx = threadIdx.x; idx < total; idx += blockDim.x) {
    int q = idx / s, j = idx - q * s;
    float lg = logits[((size_t)(b * NQ + q)) * NC + ids[j]];
    float c  = -__expf(lg - g_lse[b * NQ + q]);
    g_cost[((size_t)b * NOBJ + j) * CPAD + q] = c;
  }
}

// ---------------------------------------------------------------------------
// Kernel 3: Jonker-Volgenant assignment, one wave32 per batch (single-wave
// workgroup -> barriers are S_NOP). Column scans / argmin parallelized over
// the 32 lanes; first-index tie-break replicates np.argmin.
// ---------------------------------------------------------------------------
__global__ __launch_bounds__(32)
void k3_hungarian(const float* __restrict__ g_cost,
                  const int* __restrict__ g_sizes, int* __restrict__ g_tgt)
{
  __shared__ float vpot[NQ + 1], minv[NQ + 1], upot[NOBJ + 2];
  __shared__ int   p[NQ + 1], way[NQ + 1], used[NQ + 1];
  const int b = blockIdx.x, lane = threadIdx.x;
  const int s = g_sizes[b];
  const float* C = g_cost + (size_t)b * NOBJ * CPAD;

  for (int j = lane; j <= NQ; j += 32) { vpot[j] = 0.f; p[j] = 0; }
  for (int i = lane; i < NOBJ + 2; i += 32) upot[i] = 0.f;
  __syncthreads();

  for (int i = 1; i <= s; ++i) {
    if (lane == 0) p[0] = i;
    for (int j = lane; j <= NQ; j += 32) { minv[j] = JINF; used[j] = 0; }
    __syncthreads();
    int j0 = 0;
    while (true) {
      if (lane == 0) used[j0] = 1;
      __syncthreads();
      const int i0 = p[j0];
      const float ui0 = upot[i0];
      const float* Crow = C + (size_t)(i0 - 1) * CPAD;
      float bestv = JINF;
      int   bestj = 0x7fffffff;
      for (int j = 1 + lane; j <= NQ; j += 32) {
        if (!used[j]) {
          float cur = Crow[j - 1] - ui0 - vpot[j];
          float mj  = minv[j];
          if (cur < mj) { minv[j] = cur; way[j] = j0; mj = cur; }
          if (mj < bestv) { bestv = mj; bestj = j; }
        }
      }
      for (int off = 16; off >= 1; off >>= 1) {
        float ov = __shfl_xor(bestv, off, 32);
        int   oj = __shfl_xor(bestj, off, 32);
        if (ov < bestv || (ov == bestv && oj < bestj)) { bestv = ov; bestj = oj; }
      }
      const int   j1    = bestj;
      const float delta = bestv;
      __syncthreads();
      for (int j = lane; j <= NQ; j += 32) {
        if (used[j]) { upot[p[j]] += delta; vpot[j] -= delta; }
        else         { minv[j] -= delta; }
      }
      __syncthreads();
      j0 = j1;
      if (p[j0] == 0) break;
    }
    if (lane == 0) {       // augmenting path (serial, tiny)
      int jj = j0;
      while (jj) { int jn = way[jj]; p[jj] = p[jn]; jj = jn; }
    }
    __syncthreads();
  }
  for (int q = lane; q < NQ; q += 32) {
    int t = p[q + 1];
    g_tgt[b * NQ + q] = (t > 0) ? (t - 1) : 0;
  }
}

// ---------------------------------------------------------------------------
// Kernel 4: loss = mean over rows of (lse[row] - logit[row, tgt[row]])
// ---------------------------------------------------------------------------
__global__ __launch_bounds__(256)
void k4_loss(const float* __restrict__ logits, const float* __restrict__ g_lse,
             const int* __restrict__ g_tgt, float* __restrict__ out)
{
  __shared__ float sred[256];
  float acc = 0.f;
  for (int r = threadIdx.x; r < ROWS; r += 256) {
    int t = g_tgt[r];
    acc += g_lse[r] - logits[(size_t)r * NC + t];
  }
  sred[threadIdx.x] = acc;
  __syncthreads();
  for (int st = 128; st >= 1; st >>= 1) {
    if (threadIdx.x < st) sred[threadIdx.x] += sred[threadIdx.x + st];
    __syncthreads();
  }
  if (threadIdx.x == 0) out[0] = sred[0] / (float)ROWS;
}

// ---------------------------------------------------------------------------
extern "C" void kernel_launch(void* const* d_in, const int* in_sizes, int n_in,
                              void* d_out, int out_size, void* d_ws,
                              size_t ws_size, hipStream_t stream)
{
  const float* logits = (const float*)d_in[0];   // [32,300,8192] f32
  const int*   labels = (const int*)d_in[1];     // [32,64] int

  char* wsb = (char*)d_ws;
  float* g_lse   = (float*)(wsb);                                  // 9600 f32
  float* g_cost  = (float*)(wsb + 38400);                          // 32*64*304 f32
  int*   g_sizes = (int*)(wsb + 38400 + (size_t)BS * NOBJ * CPAD * 4);
  int*   g_tgt   = (int*)((char*)g_sizes + 128);                   // 9600 int

  k1_row_lse  <<<ROWS / 16, 256, 0, stream>>>(logits, g_lse);
  k2_cost     <<<BS,        256, 0, stream>>>(logits, labels, g_lse, g_cost, g_sizes);
  k3_hungarian<<<BS,         32, 0, stream>>>(g_cost, g_sizes, g_tgt);
  k4_loss     <<<1,         256, 0, stream>>>(logits, g_lse, g_tgt, (float*)d_out);
}
